// YoloxLoss_45045617000952
// MI455X (gfx1250) — compile-verified, matrix operand order checked by reading, not run on hardware
//
#include <hip/hip_runtime.h>
#include <hip/hip_bf16.h>
#include <math.h>

typedef __attribute__((ext_vector_type(16))) _Float16 v16h;
typedef __attribute__((ext_vector_type(8)))  float    v8f;

#define NCLS   80
#define NA     8400
#define NB     32
#define NG     30
#define A0SZ   6400
#define A1SZ   1600
#define EPSC   1e-7f

// ---------------- helpers ----------------
__device__ __forceinline__ void level_of(int a, int& al, int& W, float& stride) {
    if (a < A0SZ)              { al = a;            W = 80; stride = 8.0f;  }
    else if (a < A0SZ + A1SZ)  { al = a - A0SZ;     W = 40; stride = 16.0f; }
    else                       { al = a - A0SZ - A1SZ; W = 20; stride = 32.0f; }
}

__device__ __forceinline__ float feat_val(const float* f8, const float* f16,
                                          const float* f32, int b, int ch, int a) {
    int al, W; float s;
    level_of(a, al, W, s);
    const float* f = (a < A0SZ) ? f8 : (a < A0SZ + A1SZ ? f16 : f32);
    int HW = W * W;
    return f[((size_t)b * 85 + ch) * HW + al];
}

__device__ __forceinline__ float softplus(float x) {
    return fmaxf(x, 0.0f) + log1pf(expf(-fabsf(x)));
}
__device__ __forceinline__ float sigmoidf(float x) {
    return 1.0f / (1.0f + expf(-x));
}

__device__ __forceinline__ float block_sum256(float x, float* red, int tid) {
    red[tid] = x; __syncthreads();
    for (int s = 128; s > 0; s >>= 1) {
        if (tid < s) red[tid] += red[tid + s];
        __syncthreads();
    }
    float r = red[0]; __syncthreads();
    return r;
}

// ---------------- k0: zero accumulators ----------------
__global__ void k_zero(float* acc) {
    if (threadIdx.x < 4) acc[threadIdx.x] = 0.0f;
}

// ---------------- k1: decode ----------------
__global__ void k_decode(const float* f8, const float* f16, const float* f32,
                         float* bbox, float* objl, float* osig) {
    int b = blockIdx.x / 33;
    int a = (blockIdx.x % 33) * 256 + threadIdx.x;
    if (a >= NA) return;
    int al, W; float s;
    level_of(a, al, W, s);
    float gx = (float)(al % W), gy = (float)(al / W);
    float tx = feat_val(f8, f16, f32, b, 0, a);
    float ty = feat_val(f8, f16, f32, b, 1, a);
    float tw = feat_val(f8, f16, f32, b, 2, a);
    float th = feat_val(f8, f16, f32, b, 3, a);
    float ob = feat_val(f8, f16, f32, b, 4, a);
    size_t i = (size_t)b * NA + a;
    bbox[i * 4 + 0] = (tx + gx) * s;
    bbox[i * 4 + 1] = (ty + gy) * s;
    bbox[i * 4 + 2] = expf(tw) * s;
    bbox[i * 4 + 3] = expf(th) * s;
    objl[i] = ob;
    osig[i] = sigmoidf(ob);
}

// ---------------- k2: WMMA class reductions ----------------
// Each wave owns a tile of 16 anchors. Two chained f16 WMMAs (ones-B) per
// K-chunk give 16 row sums in f32: s_neg = sum(-log1p(-p)), sp = sum(softplus).
__global__ void k_wmma_cls(const float* f8, const float* f16, const float* f32,
                           const float* osig, float* sneg, float* spsum) {
    const int TPB = 66;            // tile-blocks per batch (66*8 waves >= 525 tiles)
    int b    = blockIdx.x / TPB;
    int blk  = blockIdx.x % TPB;
    int wave = threadIdx.x >> 5;
    int lane = threadIdx.x & 31;
    int tile = blk * 8 + wave;
    if (tile >= (NA / 16)) return;           // uniform per-wave exit
    int half = lane >> 4;                    // 0: lanes 0-15, 1: lanes 16-31
    int m    = lane & 15;
    int a0   = tile * 16;
    int a    = a0 + m;
    float os = osig[(size_t)b * NA + a];

    v16h bones;
#pragma unroll
    for (int j = 0; j < 16; ++j) bones[j] = (_Float16)1.0f;

    v8f acc_t  = {};
    v8f acc_sp = {};

#pragma unroll
    for (int chunk = 0; chunk < 3; ++chunk) {
        int cbase = chunk * 32;
        v16h at, asp;
#pragma unroll
        for (int j = 0; j < 16; ++j) {
            int v = j >> 1, o = j & 1;
            int K = (v < 4) ? (half * 8 + 2 * v + o)
                            : (16 + half * 8 + 2 * (v - 4) + o);
            int c = cbase + K;
            float t = 0.0f, sp = 0.0f;
            if (c < NCLS) {
                float x  = feat_val(f8, f16, f32, b, 5 + c, a);
                sp       = softplus(x);
                float p  = sqrtf(sigmoidf(x) * os);
                p        = fminf(fmaxf(p, EPSC), 1.0f - EPSC);
                t        = -log1pf(-p);
            }
            at[j]  = (_Float16)t;
            asp[j] = (_Float16)sp;
        }
        acc_t  = __builtin_amdgcn_wmma_f32_16x16x32_f16(
                     false, at,  false, bones, (short)0, acc_t,  false, false);
        acc_sp = __builtin_amdgcn_wmma_f32_16x16x32_f16(
                     false, asp, false, bones, (short)0, acc_sp, false, false);
    }
    // D layout: VGPR j, lane 0 -> row j (N=0); lane 16 -> row 8+j (N=0)
    if (m == 0) {
#pragma unroll
        for (int j = 0; j < 8; ++j) {
            int row = half * 8 + j;
            sneg [(size_t)b * NA + a0 + row] = acc_t[j];
            spsum[(size_t)b * NA + a0 + row] = acc_sp[j];
        }
    }
}

// ---------------- k3: anchor candidate mask ----------------
__global__ void k_cand(const float* labels, float* cand) {
    int b = blockIdx.x / 33;
    int a = (blockIdx.x % 33) * 256 + threadIdx.x;
    if (a >= NA) return;
    int al, W; float s;
    level_of(a, al, W, s);
    float xc = ((float)(al % W) + 0.5f) * s;
    float yc = ((float)(al / W) + 0.5f) * s;
    float r  = 2.5f * s;
    bool any = false;
    for (int g = 0; g < NG; ++g) {
        const float* L = labels + ((size_t)b * NG + g) * 5;
        float gx = L[0], gy = L[1], gw = L[2], gh = L[3];
        bool ib = (xc > gx - gw * 0.5f) && (xc < gx + gw * 0.5f) &&
                  (yc > gy - gh * 0.5f) && (yc < gy + gh * 0.5f);
        bool ic = (fabsf(xc - gx) < r) && (fabsf(yc - gy) < r);
        any = any || ib || ic;
    }
    cand[(size_t)b * NA + a] = any ? 1.0f : 0.0f;
}

// ---------------- k4: cost & pair-IoU matrices ----------------
__global__ void k_cost(const float* f8, const float* f16, const float* f32,
                       const float* labels, const float* bbox, const float* osig,
                       const float* sneg, const float* cand,
                       float* cost, float* piou) {
    int bid = blockIdx.x;
    int b = bid / (NG * 33);
    int r = bid % (NG * 33);
    int g = r / 33;
    int a = (r % 33) * 256 + threadIdx.x;
    if (a >= NA) return;

    const float* L = labels + ((size_t)b * NG + g) * 5;
    float gx = L[0], gy = L[1], gw = L[2], gh = L[3];
    int gcls = (int)L[4];

    const float* pb = bbox + ((size_t)b * NA + a) * 4;
    __builtin_prefetch(pb + 256 * 4, 0, 1);
    float bx = pb[0], by = pb[1], bw = pb[2], bh = pb[3];

    // pair IoU
    float tlx = fmaxf(gx - gw * 0.5f, bx - bw * 0.5f);
    float tly = fmaxf(gy - gh * 0.5f, by - bh * 0.5f);
    float brx = fminf(gx + gw * 0.5f, bx + bw * 0.5f);
    float bry = fminf(gy + gh * 0.5f, by + bh * 0.5f);
    float en  = ((tlx < brx) && (tly < bry)) ? 1.0f : 0.0f;
    float inter = (brx - tlx) * (bry - tly) * en;
    float iou = inter / (gw * gh + bw * bh - inter + 1e-16f);

    // geometric masks
    int al, W; float s;
    level_of(a, al, W, s);
    float xc = ((float)(al % W) + 0.5f) * s;
    float yc = ((float)(al / W) + 0.5f) * s;
    float rr = 2.5f * s;
    bool ib = (xc > gx - gw * 0.5f) && (xc < gx + gw * 0.5f) &&
              (yc > gy - gh * 0.5f) && (yc < gy + gh * 0.5f);
    bool ic = (fabsf(xc - gx) < rr) && (fabsf(yc - gy) < rr);
    bool in_both = ib && ic;
    float cd = cand[(size_t)b * NA + a];

    // cls cost: recompute p only at the GT class
    float x  = feat_val(f8, f16, f32, b, 5 + gcls, a);
    float p  = sqrtf(sigmoidf(x) * osig[(size_t)b * NA + a]);
    p        = fminf(fmaxf(p, EPSC), 1.0f - EPSC);
    float cls_cost = sneg[(size_t)b * NA + a] - logf(p) + log1pf(-p);
    float iou_cost = -logf(iou + 1e-8f);

    float c = cls_cost + 3.0f * iou_cost
            + (in_both ? 0.0f : 100000.0f)
            + (cd > 0.5f ? 0.0f : 1000000.0f);

    size_t idx = ((size_t)b * NG + g) * NA + a;
    cost[idx] = c;
    piou[idx] = iou;
}

// ---------------- k5: per-(b,g) dynamic-k threshold ----------------
__global__ void k_topk(const float* cost, const float* piou, const float* cand,
                       float* thresh) {
    int b = blockIdx.x / NG;
    int g = blockIdx.x % NG;
    int tid = threadIdx.x;
    const float* crow = cost + ((size_t)b * NG + g) * NA;
    const float* irow = piou + ((size_t)b * NG + g) * NA;
    const float* cb   = cand + (size_t)b * NA;

    float iouv[10], costv[10];
#pragma unroll
    for (int i = 0; i < 10; ++i) { iouv[i] = -1e30f; costv[i] = 1e30f; }

    for (int a = tid; a < NA; a += 256) {
        float ic = irow[a] * cb[a];
        if (ic > iouv[9]) {
            int i = 9;
            while (i > 0 && iouv[i - 1] < ic) { iouv[i] = iouv[i - 1]; --i; }
            iouv[i] = ic;
        }
        float cc = crow[a];
        if (cc < costv[9]) {
            int i = 9;
            while (i > 0 && costv[i - 1] > cc) { costv[i] = costv[i - 1]; --i; }
            costv[i] = cc;
        }
    }

    __shared__ float sval[256];
    __shared__ int   sidx[256];
    __shared__ float ksmall[10];
    __shared__ float ssum;

    // 10 rounds of block-argmax over candidate IoUs
    int pi = 0;
    float sum0 = 0.0f;
    for (int r = 0; r < 10; ++r) {
        sval[tid] = (pi < 10) ? iouv[pi] : -1e30f;
        sidx[tid] = tid;
        __syncthreads();
        for (int s = 128; s > 0; s >>= 1) {
            if (tid < s && sval[tid + s] > sval[tid]) {
                sval[tid] = sval[tid + s]; sidx[tid] = sidx[tid + s];
            }
            __syncthreads();
        }
        if (tid == 0) sum0 += fmaxf(sval[0], 0.0f);
        int w = sidx[0];
        __syncthreads();
        if (w == tid) ++pi;
    }
    if (tid == 0) ssum = sum0;

    // 10 rounds of block-argmin over costs
    pi = 0;
    for (int r = 0; r < 10; ++r) {
        sval[tid] = (pi < 10) ? costv[pi] : 1e30f;
        sidx[tid] = tid;
        __syncthreads();
        for (int s = 128; s > 0; s >>= 1) {
            if (tid < s && sval[tid + s] < sval[tid]) {
                sval[tid] = sval[tid + s]; sidx[tid] = sidx[tid + s];
            }
            __syncthreads();
        }
        if (tid == 0) ksmall[r] = sval[0];
        int w = sidx[0];
        __syncthreads();
        if (w == tid) ++pi;
    }
    __syncthreads();
    if (tid == 0) {
        int dk = (int)ssum;
        dk = (dk < 1) ? 1 : (dk > 10 ? 10 : dk);
        thresh[(size_t)b * NG + g] = ksmall[dk - 1];
    }
}

// ---------------- k6: resolve matches + loss partials ----------------
__global__ void k_resolve(const float* f8, const float* f16, const float* f32,
                          const float* labels, const float* bbox, const float* objl,
                          const float* spsum, const float* cost, const float* piouM,
                          const float* thresh, float* acc) {
    __shared__ float red[256];
    int b = blockIdx.x / 33;
    int a = (blockIdx.x % 33) * 256 + threadIdx.x;
    bool valid = (a < NA);
    int aa = valid ? a : 0;

    int cnt = 0, first = -1, cmin = 0;
    float piou_first = 0.0f, cmin_val = 1e30f;
    for (int g = 0; g < NG; ++g) {
        size_t idx = ((size_t)b * NG + g) * NA + aa;
        float c  = cost[idx];
        float io = piouM[idx];
        if (c < cmin_val) { cmin_val = c; cmin = g; }
        if (c <= thresh[(size_t)b * NG + g]) {
            ++cnt;
            if (first < 0) { first = g; piou_first = io; }
        }
    }
    float fgf = (valid && cnt > 0) ? 1.0f : 0.0f;
    int mg; float pio;
    if (cnt > 1)      { mg = cmin;  pio = piouM[((size_t)b * NG + cmin) * NA + aa]; }
    else if (cnt == 1){ mg = first; pio = piou_first; }
    else              { mg = 0;     pio = 0.0f; }

    const float* L = labels + ((size_t)b * NG + mg) * 5;
    float gx = L[0], gy = L[1], gw = L[2], gh = L[3];
    int   mcls = (int)L[4];

    const float* pb = bbox + ((size_t)b * NA + aa) * 4;
    float bx = pb[0], by = pb[1], bw = pb[2], bh = pb[3];

    // GIoU
    float tlx = fmaxf(bx - bw * 0.5f, gx - gw * 0.5f);
    float tly = fmaxf(by - bh * 0.5f, gy - gh * 0.5f);
    float brx = fminf(bx + bw * 0.5f, gx + gw * 0.5f);
    float bry = fminf(by + bh * 0.5f, gy + gh * 0.5f);
    float en  = ((tlx < brx) && (tly < bry)) ? 1.0f : 0.0f;
    float inter = (brx - tlx) * (bry - tly) * en;
    float uni   = bw * bh + gw * gh - inter;
    float iou   = inter / (uni + 1e-16f);
    float ctlx = fminf(bx - bw * 0.5f, gx - gw * 0.5f);
    float ctly = fminf(by - bh * 0.5f, gy - gh * 0.5f);
    float cbrx = fmaxf(bx + bw * 0.5f, gx + gw * 0.5f);
    float cbry = fmaxf(by + bh * 0.5f, gy + gh * 0.5f);
    float area_c = (cbrx - ctlx) * (cbry - ctly);
    float giou = iou - (area_c - uni) / fmaxf(area_c, 1e-16f);
    float liou = (1.0f - fminf(fmaxf(giou, -1.0f), 1.0f)) * fgf;

    float ob   = objl[(size_t)b * NA + aa];
    float lobj = valid ? (softplus(ob) - ob * fgf) : 0.0f;

    float xmc  = feat_val(f8, f16, f32, b, 5 + mcls, aa);
    float lcls = (spsum[(size_t)b * NA + aa] - xmc * pio) * fgf;

    float s0 = block_sum256(liou, red, threadIdx.x);
    float s1 = block_sum256(lobj, red, threadIdx.x);
    float s2 = block_sum256(lcls, red, threadIdx.x);
    float s3 = block_sum256(fgf,  red, threadIdx.x);
    if (threadIdx.x == 0) {
        atomicAdd(&acc[0], s0);
        atomicAdd(&acc[1], s1);
        atomicAdd(&acc[2], s2);
        atomicAdd(&acc[3], s3);
    }
}

// ---------------- k7: finalize ----------------
__global__ void k_final(const float* acc, float* out) {
    if (threadIdx.x == 0) {
        float nfg = fmaxf(acc[3], 1.0f);
        out[0] = (5.0f * acc[0] + acc[1] + acc[2]) / nfg;
    }
}

// ---------------- launch ----------------
extern "C" void kernel_launch(void* const* d_in, const int* in_sizes, int n_in,
                              void* d_out, int out_size, void* d_ws, size_t ws_size,
                              hipStream_t stream) {
    const float* f8  = (const float*)d_in[0];
    const float* f16 = (const float*)d_in[1];
    const float* f32 = (const float*)d_in[2];
    const float* lab = (const float*)d_in[3];
    float* out = (float*)d_out;
    float* w   = (float*)d_ws;

    const size_t nBA = (size_t)NB * NA;
    float* bbox   = w;                         // NB*NA*4
    float* objl   = bbox   + nBA * 4;          // NB*NA
    float* osig   = objl   + nBA;
    float* sneg   = osig   + nBA;
    float* spsum  = sneg   + nBA;
    float* cand   = spsum  + nBA;
    float* cost   = cand   + nBA;              // NB*NG*NA
    float* piou   = cost   + (size_t)NB * NG * NA;
    float* thresh = piou   + (size_t)NB * NG * NA;  // NB*NG
    float* acc    = thresh + (size_t)NB * NG;       // 4

    k_zero   <<<1, 32, 0, stream>>>(acc);
    k_decode <<<NB * 33, 256, 0, stream>>>(f8, f16, f32, bbox, objl, osig);
    k_wmma_cls<<<NB * 66, 256, 0, stream>>>(f8, f16, f32, osig, sneg, spsum);
    k_cand   <<<NB * 33, 256, 0, stream>>>(lab, cand);
    k_cost   <<<NB * NG * 33, 256, 0, stream>>>(f8, f16, f32, lab, bbox, osig,
                                                sneg, cand, cost, piou);
    k_topk   <<<NB * NG, 256, 0, stream>>>(cost, piou, cand, thresh);
    k_resolve<<<NB * 33, 256, 0, stream>>>(f8, f16, f32, lab, bbox, objl,
                                           spsum, cost, piou, thresh, acc);
    k_final  <<<1, 32, 0, stream>>>(acc, out);
}
